// TensorNetworkMERA_66159676227589
// MI455X (gfx1250) — matrix operand order repaired and checked by reading, not compiled, not run on hardware
//
#include <hip/hip_runtime.h>

// ---------------------------------------------------------------------------
// TensorNetworkMERA on gfx1250 (MI455X, wave32, WMMA).
//
// Strategy:
//  * Pre-contract u over the mean axes: U1[i,j,k]=mean_l u, U2[i,j,l]=mean_k u.
//    Each layer then is: x = s0 (x) s1 (K=256), o1 = x@U1, o2 = x@U2,
//    y = o1 (x) o2 (K=256), out = y@wflat.   (8x fewer FLOPs than naive comb.)
//  * All GEMMs via v_wmma_f32_16x16x32_bf16 (bf16 in, f32 accumulate):
//    16 pairs per wave = one 16-row WMMA tile, K=256 = 8 chunks of 32.
//  * Embedding GEMM (262144 x 128 x 16) staged through LDS, coalesced; this is
//    the only HBM-bound stage (134 MB read ~ 6 us at 23.3 TB/s). Sites (16 MB)
//    stay L2-resident across the 6 layer kernels.
//  * B matrices pre-swizzled into WMMA lane order so each lane loads its v16bf
//    with one 32B load.
// ---------------------------------------------------------------------------

typedef __bf16 bf16;
typedef __attribute__((ext_vector_type(16))) __bf16 v16bf;
typedef __attribute__((ext_vector_type(8)))  float  v8f;

#define WMMA_BF16(A, B, C) \
  __builtin_amdgcn_wmma_f32_16x16x32_bf16(false, (A), false, (B), (short)0, (C), false, false)

// ---------------------------------------------------------------------------
// Prep: bf16-swizzle W_emb and per-layer U1/U2/wflat into WMMA B-operand order.
// B layout (16-bit, 32x16): lane<16 holds K=0..7,16..23, lane>=16 holds K=8..15,24..31,
// N = lane&15. We store so that lane (half,nl) chunk c reads 16 contiguous
// bf16: addr = ((c*2+half)*16 + nl)*16 + e, value = B[K=32c+8*half+(e&7)+16*(e>>3)][N=nl].
// ---------------------------------------------------------------------------
__global__ void prep_kernel(const float* __restrict__ W_emb,
                            const float* __restrict__ u,
                            const float* __restrict__ w,
                            bf16* __restrict__ Wp,   // 2048  (K=128, N=16)
                            bf16* __restrict__ Lp)   // 6 * (U1|U2|WF) * 4096
{
  int idx = blockIdx.x * blockDim.x + threadIdx.x;
  if (idx >= 2048 + 6 * 3 * 4096) return;
  if (idx < 2048) {
    int e = idx & 15, nl = (idx >> 4) & 15, half = (idx >> 8) & 1, c = idx >> 9;
    int K = 32 * c + 8 * half + (e & 7) + 16 * (e >> 3);        // K < 128
    Wp[idx] = (bf16)W_emb[K * 16 + nl];
  } else {
    int t2 = idx - 2048;
    int layer = t2 / 12288;
    int r = t2 % 12288;
    int which = r / 4096;                  // 0=U1, 1=U2, 2=wflat
    int t = r % 4096;
    int e = t & 15, nl = (t >> 4) & 15, half = (t >> 8) & 1, c = t >> 9; // c<8
    int K = 32 * c + 8 * half + (e & 7) + 16 * (e >> 3);        // K = i*16+j < 256
    int i = K >> 4, j = K & 15;
    const float* ul = u + layer * 65536;   // u[l][i][j][k][l2]
    const float* wl = w + layer * 4096;    // w[l][a][i][j]
    float val;
    if (which == 0) {                      // U1[(i,j)][k=nl] = mean_l2 u
      float s = 0.f;
      for (int l2 = 0; l2 < 16; ++l2) s += ul[i * 4096 + j * 256 + nl * 16 + l2];
      val = s * (1.f / 16.f);
    } else if (which == 1) {               // U2[(i,j)][l2=nl] = mean_k u
      float s = 0.f;
      for (int k = 0; k < 16; ++k) s += ul[i * 4096 + j * 256 + k * 16 + nl];
      val = s * (1.f / 16.f);
    } else {                               // wflat[(i,j)][a=nl]
      val = wl[nl * 256 + K];
    }
    Lp[layer * 12288 + which * 4096 + t] = (bf16)val;
  }
}

// ---------------------------------------------------------------------------
// Embedding: sites = normalize(seq @ W_emb + b). One wave per 16-row tile.
// 16x128 fp32 tile staged through LDS with fully coalesced 512B loads.
// ---------------------------------------------------------------------------
__global__ void __launch_bounds__(128) embed_kernel(const float* __restrict__ seq,
                                                    const bf16*  __restrict__ Wp,
                                                    const float* __restrict__ bias,
                                                    float* __restrict__ sites)
{
  __shared__ float smem[4][16 * 132];      // 132-float row pitch: no bank conflicts
  int lane = threadIdx.x & 31, wid = threadIdx.x >> 5;
  long tile = (long)blockIdx.x * 4 + wid;
  long rowbase = tile * 16;
  const float* src = seq + rowbase * 128;
  float* L = smem[wid];
  #pragma unroll
  for (int r = 0; r < 16; ++r) {           // coalesced: 32 lanes x float4 = one row
    float4 v = ((const float4*)(src + (long)r * 128))[lane];
    *(float4*)(L + r * 132 + lane * 4) = v;
  }
  __syncthreads();

  int nl = lane & 15, half = lane >> 4, off = half * 8;
  const float* myrow = L + nl * 132;       // A-row for this lane
  const v16bf* Bp = (const v16bf*)Wp;
  v8f acc = {};
  #pragma unroll
  for (int c = 0; c < 4; ++c) {            // K = 128 in 4 chunks of 32
    v16bf A;
    #pragma unroll
    for (int e = 0; e < 8; ++e) {
      A[e]     = (bf16)myrow[32 * c + off + e];        // K = 32c+off+e
      A[e + 8] = (bf16)myrow[32 * c + 16 + off + e];   // K = 32c+16+off+e
    }
    v16bf B = Bp[(c * 2 + half) * 16 + nl];
    acc = WMMA_BF16(A, B, acc);
  }
  // bias + row-normalize: C/D layout M = v + 8*half, N = nl.
  float bv = bias[nl];
  float vals[8], ss[8];
  #pragma unroll
  for (int v = 0; v < 8; ++v) { vals[v] = acc[v] + bv; ss[v] = vals[v] * vals[v]; }
  #pragma unroll
  for (int m = 1; m <= 8; m <<= 1) {       // reduce across the 16 N-lanes
    #pragma unroll
    for (int v = 0; v < 8; ++v) ss[v] += __shfl_xor(ss[v], m, 32);
  }
  #pragma unroll
  for (int v = 0; v < 8; ++v) {
    float nrm = fmaxf(sqrtf(ss[v]), 1e-12f);
    sites[(rowbase + v + 8 * half) * 16 + nl] = vals[v] / nrm;
  }
}

// ---------------------------------------------------------------------------
// One MERA layer. One wave per 16-pair tile:
//   GEMM1: A = s0 (x) s1  (16 x 256)  ->  o1 (B=U1), o2 (B=U2)
//   LDS transpose (C layout is N-across-lanes, we need rows-across-lanes)
//   GEMM2: A = o1 (x) o2  (16 x 256)  ->  out (B=wflat)
// ---------------------------------------------------------------------------
__global__ void __launch_bounds__(128) layer_kernel(const float* __restrict__ in,
                                                    float* __restrict__ out,
                                                    const bf16* __restrict__ Lp)
{
  __shared__ float smem[4][1104];          // 32x17 input + 2 x 16x17 o1/o2
  int lane = threadIdx.x & 31, wid = threadIdx.x >> 5;
  int tile = blockIdx.x * 4 + wid;
  long pairbase = (long)tile * 16;
  float* L = smem[wid];
  const float* src = in + pairbase * 32;   // 32 site rows of 16 floats
  #pragma unroll
  for (int it = 0; it < 16; ++it) {        // coalesced scalar loads, padded rows
    int fi = it * 32 + lane;               // 0..511
    L[(fi >> 4) * 17 + (fi & 15)] = src[fi];
  }
  __syncthreads();

  int nl = lane & 15, half = lane >> 4, jb = half * 8;
  const float* s0p = L + (2 * nl) * 17;
  const float* s1p = L + (2 * nl + 1) * 17;
  float s1v[8];
  #pragma unroll
  for (int e = 0; e < 8; ++e) s1v[e] = s1p[jb + e];

  const v16bf* U1v = (const v16bf*)Lp;
  const v16bf* U2v = (const v16bf*)(Lp + 4096);
  const v16bf* WFv = (const v16bf*)(Lp + 8192);

  v8f c0 = {}, c1 = {};
  #pragma unroll
  for (int c = 0; c < 8; ++c) {            // K = 256 in 8 chunks of 32
    float a0 = s0p[2 * c], a1 = s0p[2 * c + 1];
    v16bf A;
    #pragma unroll
    for (int e = 0; e < 8; ++e) {
      A[e]     = (bf16)(a0 * s1v[e]);      // x[i=2c  ][j=jb+e]
      A[e + 8] = (bf16)(a1 * s1v[e]);      // x[i=2c+1][j=jb+e]
    }
    c0 = WMMA_BF16(A, U1v[(c * 2 + half) * 16 + nl], c0);
    c1 = WMMA_BF16(A, U2v[(c * 2 + half) * 16 + nl], c1);
  }

  __syncthreads();
  #pragma unroll
  for (int v = 0; v < 8; ++v) {             // transpose o1/o2 via LDS
    L[544 + (v + 8 * half) * 17 + nl] = c0[v];
    L[816 + (v + 8 * half) * 17 + nl] = c1[v];
  }
  __syncthreads();

  const float* o1p = L + 544 + nl * 17;
  float o2v[8];
  #pragma unroll
  for (int e = 0; e < 8; ++e) o2v[e] = L[816 + nl * 17 + jb + e];

  v8f c2 = {};
  #pragma unroll
  for (int c = 0; c < 8; ++c) {
    float a0 = o1p[2 * c], a1 = o1p[2 * c + 1];
    v16bf A;
    #pragma unroll
    for (int e = 0; e < 8; ++e) {
      A[e]     = (bf16)(a0 * o2v[e]);
      A[e + 8] = (bf16)(a1 * o2v[e]);
    }
    c2 = WMMA_BF16(A, WFv[(c * 2 + half) * 16 + nl], c2);
  }
  #pragma unroll
  for (int v = 0; v < 8; ++v)
    out[(pairbase + v + 8 * half) * 16 + nl] = c2[v];
}

// ---------------------------------------------------------------------------
// phi_q tail: per-batch 16x16 + two 8x8 symmetric Jacobi eigensolves of
// corr*corr^T (singular values squared). One thread per batch (128 total).
// ---------------------------------------------------------------------------
__device__ float ent_entropy(const float* A, const float* B, int n) {
  float C[16][16], M[16][16];
  for (int i = 0; i < n; ++i)
    for (int j = 0; j < n; ++j) {
      float s = 0.f;
      for (int k = 0; k < 16; ++k) s += A[i * 16 + k] * B[j * 16 + k];
      C[i][j] = s;
    }
  for (int i = 0; i < n; ++i)
    for (int j = 0; j <= i; ++j) {
      float s = 0.f;
      for (int t = 0; t < n; ++t) s += C[i][t] * C[j][t];
      M[i][j] = s; M[j][i] = s;
    }
  for (int sweep = 0; sweep < 12; ++sweep)
    for (int p = 0; p < n - 1; ++p)
      for (int q = p + 1; q < n; ++q) {
        float apq = M[p][q];
        if (fabsf(apq) < 1e-20f) continue;
        float app = M[p][p], aqq = M[q][q];
        float tau = (aqq - app) / (2.f * apq);
        float t = (tau >= 0.f ? 1.f : -1.f) / (fabsf(tau) + sqrtf(1.f + tau * tau));
        float cth = 1.f / sqrtf(1.f + t * t);
        float sth = t * cth;
        for (int k = 0; k < n; ++k) {
          float akp = M[k][p], akq = M[k][q];
          M[k][p] = cth * akp - sth * akq;
          M[k][q] = sth * akp + cth * akq;
        }
        for (int k = 0; k < n; ++k) {
          float apk = M[p][k], aqk = M[q][k];
          M[p][k] = cth * apk - sth * aqk;
          M[q][k] = sth * apk + cth * aqk;
        }
      }
  float s[16], sum = 0.f;
  for (int i = 0; i < n; ++i) { s[i] = sqrtf(fmaxf(M[i][i], 0.f)); sum += s[i]; }
  float inv = 1.f / (sum + 1e-8f);
  float S = 0.f;
  for (int i = 0; i < n; ++i) { float sn = s[i] * inv; S -= sn * logf(sn + 1e-8f); }
  return S;
}

__global__ void phi_kernel(const float* __restrict__ sites, float* __restrict__ phi) {
  int b = blockIdx.x * blockDim.x + threadIdx.x;
  if (b >= 128) return;
  const float* st = sites + b * 512;                 // (32 sites x 16)
  float Sint = ent_entropy(st,        st + 256, 16); // rows 0..15 vs 16..31
  float S1   = ent_entropy(st,        st + 128,  8); // rows 0..7  vs 8..15
  float S2   = ent_entropy(st + 256,  st + 384,  8); // rows 16..23 vs 24..31
  phi[b] = fmaxf(Sint - (S1 + S2), 0.f);
}

// ---------------------------------------------------------------------------
extern "C" void kernel_launch(void* const* d_in, const int* in_sizes, int n_in,
                              void* d_out, int out_size, void* d_ws, size_t ws_size,
                              hipStream_t stream) {
  const float* seq  = (const float*)d_in[0];  // (128,2048,128)
  const float* Wemb = (const float*)d_in[1];  // (128,16)
  const float* bemb = (const float*)d_in[2];  // (16,)
  const float* u    = (const float*)d_in[3];  // (6,16,16,16,16)
  const float* w    = (const float*)d_in[4];  // (6,16,16,16)

  char* ws = (char*)d_ws;
  bf16*  Wp = (bf16*)ws;                         // 4 KB
  bf16*  Lp = (bf16*)(ws + 4096);                // 144 KB
  float* sA = (float*)(ws + (1 << 18));          // 16 MB (n=2048 sites)
  float* sB = (float*)(ws + (1 << 18) + (16u << 20)); // 8 MB
  float* outF = (float*)d_out;                   // 65536 sites + 128 phi

  prep_kernel<<<296, 256, 0, stream>>>(Wemb, u, w, Wp, Lp);
  embed_kernel<<<4096, 128, 0, stream>>>(seq, Wp, bemb, sA);   // 16384 tiles

  // pairs per layer: 131072, 65536, 32768, 16384, 8192, 4096  (grid = pairs/64)
  layer_kernel<<<2048, 128, 0, stream>>>(sA, sB,   Lp + 0 * 12288);
  layer_kernel<<<1024, 128, 0, stream>>>(sB, sA,   Lp + 1 * 12288);
  layer_kernel<<< 512, 128, 0, stream>>>(sA, sB,   Lp + 2 * 12288);
  layer_kernel<<< 256, 128, 0, stream>>>(sB, sA,   Lp + 3 * 12288);
  layer_kernel<<< 128, 128, 0, stream>>>(sA, sB,   Lp + 4 * 12288);
  layer_kernel<<<  64, 128, 0, stream>>>(sB, outF, Lp + 5 * 12288);

  phi_kernel<<<1, 128, 0, stream>>>(outF, outF + 65536);
}
